// GCN_25580825215277
// MI455X (gfx1250) — matrix operand order repaired and checked by reading, not compiled, not run on hardware
//
#include <hip/hip_runtime.h>
#include <math.h>
#include <stdint.h>

// Problem constants (from reference)
#define NTOT   131072   // B*N nodes
#define BGR    1024     // graphs
#define NNODE  128      // nodes per graph
#define FIN0   64
#define HID    128
#define EPG    1024     // edges per graph
#define EDG    (BGR*EPG)
#define EPSBN  1e-5f

typedef __attribute__((ext_vector_type(16))) __bf16        v16bf;
typedef __attribute__((ext_vector_type(8)))  float         v8f;
typedef __attribute__((ext_vector_type(4)))  unsigned int  v4u;
typedef __attribute__((ext_vector_type(8)))  int           v8i;
typedef __attribute__((ext_vector_type(4)))  int           v4i;

// The gfx1250 TDM header ships only with the toolchain whose
// __builtin_amdgcn_tensor_load_to_lds takes 6 args; ROCm 7.2 takes 5.
#if __has_include(<hip/amd_detail/amd_gfx1250_TDM.h>)
#define TDM_ARGS6 1
#else
#define TDM_ARGS6 0
#endif

static __device__ __forceinline__ __bf16 f2bf(float f) {
  unsigned int u = __builtin_bit_cast(unsigned int, f);
  unsigned short s = (unsigned short)(u >> 16);   // truncate-to-bf16
  return __builtin_bit_cast(__bf16, s);
}

static __device__ __forceinline__ v8f wmma_bf16(v16bf a, v16bf b, v8f c) {
  return __builtin_amdgcn_wmma_f32_16x16x32_bf16(false, a, false, b, (short)0, c,
                                                 false, false);
}

// ------------- Tensor Data Mover: 2D f32 tile -> LDS, +2 dword row padding -----
// D# per CDNA5 ISA. tensor == tile (no OOB), data_size=4B,
// pad_enable with pad_amount=2 dwords so LDS row stride = row_elems+2
// (stride mod 64 banks == 2 -> conflict-free fragment reads).
static __device__ __forceinline__ void tdm_load_tile_2d(
    unsigned lds_addr, const void* gptr, unsigned row_elems, unsigned nrows,
    unsigned pad_interval_code) {
  unsigned long long ga = (unsigned long long)(uintptr_t)gptr;
  v4u g0;
  g0[0] = 1u;                                               // count=1, user mode
  g0[1] = lds_addr;                                         // LDS byte address
  g0[2] = (unsigned)ga;                                     // global_addr[31:0]
  g0[3] = (unsigned)((ga >> 32) & 0x01FFFFFFu) | (2u << 30); // ga[56:32] | type=2
  v8i g1;
  g1[0] = (int)((2u << 16)                       // data_size = 4 bytes
              | (1u << 20)                       // pad_enable
              | (pad_interval_code << 22)        // 5 -> every 64 dw, 6 -> every 128 dw
              | (1u << 25));                     // pad_amount code 1 = 2 dwords
  g1[1] = (int)((row_elems & 0xFFFFu) << 16);                   // tensor_dim0 lo
  g1[2] = (int)((row_elems >> 16) | ((nrows & 0xFFFFu) << 16)); // dim0 hi|dim1 lo
  g1[3] = (int)((nrows >> 16) | (row_elems << 16));             // dim1 hi|tile_dim0
  g1[4] = (int)(nrows & 0xFFFFu);                               // tile_dim1, tile_dim2=0
  g1[5] = (int)row_elems;                                       // dim0_stride lo32
  g1[6] = 0;
  g1[7] = 0;
  v4i gz = {0, 0, 0, 0};
#if TDM_ARGS6
  v8i gz8 = {0, 0, 0, 0, 0, 0, 0, 0};
  __builtin_amdgcn_tensor_load_to_lds(g0, g1, gz, gz, gz8, 0);
#else
  __builtin_amdgcn_tensor_load_to_lds(g0, g1, gz, gz, 0);
#endif
}

// B fragment (32x16 bf16) from pre-packed fragment-major buffer:
// frag index f = kt*(N/16) + nt ; lane-contiguous 16 bf16 (32B) per lane.
static __device__ __forceinline__ v16bf load_b_packed(const __bf16* __restrict__ WP,
                                                      int frag, int lane) {
  return *(const v16bf*)(WP + (((size_t)frag * 32) + lane) * 16);
}

// A fragment (16x32 bf16, MxK) straight from global f32 row-major A[.][lda]
static __device__ __forceinline__ v16bf load_a_frag_g(const float* __restrict__ A,
                                                      int lda, int row0, int k0) {
  int lane = threadIdx.x & 31;
  int m = lane & 15, h = lane >> 4;
  const float* src = A + (long)(row0 + m) * lda + k0;
  v16bf a;
#pragma unroll
  for (int e = 0; e < 8; ++e) {
    a[e]     = f2bf(src[8 * h + e]);
    a[8 + e] = f2bf(src[16 + 8 * h + e]);
  }
  return a;
}

// ---------------- utility fills ----------------
__global__ void fill_f32(float* p, float v, long n) {
  long i = blockIdx.x * 256L + threadIdx.x;
  if (i < n) p[i] = v;
}
__global__ void fill_i32(int* p, int v, long n) {
  long i = blockIdx.x * 256L + threadIdx.x;
  if (i < n) p[i] = v;
}

// ---------------- pack f32 weights into fragment-major bf16 ----------------
// One thread per (fragment, lane): writes 16 contiguous bf16 (32B).
__global__ void pack_b_kernel(const float* __restrict__ W, int K, int N,
                              __bf16* __restrict__ out) {
  const int idx = blockIdx.x * blockDim.x + threadIdx.x;
  const int nfr = (K >> 5) * (N >> 4);
  if (idx >= nfr * 32) return;
  const int frag = idx >> 5, lane = idx & 31;
  const int kt = frag / (N >> 4), nt = frag % (N >> 4);
  const int n = nt * 16 + (lane & 15), h = lane >> 4;
  __bf16* o = out + (size_t)idx * 16;
#pragma unroll
  for (int e = 0; e < 16; ++e)
    o[e] = f2bf(W[(size_t)(kt * 32 + 8 * h + e + (e >= 8 ? 8 : 0)) * N + n]);
}

// ---------------- deterministic per-graph edge aggregation ----------------
// One block per graph; fixed edge order -> bitwise deterministic (in-order DS
// RMW per wave, disjoint feature lanes across waves). No fp atomics.
__global__ __launch_bounds__(128) void gather_kernel(
    const float* __restrict__ hin, const int* __restrict__ src,
    const int* __restrict__ dst, const int* __restrict__ emask,
    int fin, float* __restrict__ agg) {
  extern __shared__ float sAgg[];          // NNODE * fin floats
  const int g = blockIdx.x;
  const int t = threadIdx.x;
  for (int i = t; i < NNODE * fin; i += 128) sAgg[i] = 0.f;
  __syncthreads();
  const int ebase = g * EPG;
  for (int e = 0; e < EPG; ++e) {
    const int eg = ebase + e;
    if (e + 8 < EPG && t < fin)
      __builtin_prefetch(hin + (long)src[ebase + e + 8] * fin + t, 0, 1);
    if (emask[eg] && t < fin) {
      const int s = src[eg];
      const int d = dst[eg] - g * NNODE;
      sAgg[d * fin + t] += hin[(long)s * fin + t];
    }
  }
  __syncthreads();
  float* out = agg + (long)g * NNODE * fin;
  for (int i = t; i < NNODE * fin; i += 128) out[i] = sAgg[i];
}

// ---------------- fused conv stage: relu/BN(agg@Wrel + b + x@Wroot) ----------------
// Block = 128 threads = 4 waves, one 16-row M tile; wave w covers N=[32w,32w+32).
// A tiles (agg & x, 16 x fin f32) are DMA'd by the TDM into LDS with 2-dword row
// padding; B comes from the pre-packed bf16 fragment buffer (b128 loads).
__global__ __launch_bounds__(128) void stage_gemm_kernel(
    const float* __restrict__ agg, const float* __restrict__ xin, int fin,
    const __bf16* __restrict__ WrelP, const __bf16* __restrict__ WrootP,
    const float* __restrict__ brel, const float* __restrict__ gamma,
    const float* __restrict__ beta, const float* __restrict__ mean,
    const float* __restrict__ var, float* __restrict__ hout) {
  __shared__ float sT[2 * 16 * (HID + 2)];     // two padded tiles (max fin=128)
  const int stride = fin + 2;                  // TDM pad: +2 dwords per row
  const int row0 = blockIdx.x * 16;
  const int wave = threadIdx.x >> 5;
  const int lane = threadIdx.x & 31;
  const int m = lane & 15, h = lane >> 4;
  const int n0 = wave * 32;
  float* sA = sT;
  float* sX = sT + 16 * stride;

  if (wave == 0) {
    const unsigned padc = (fin == HID) ? 6u : 5u;   // pad every 128 / 64 dwords
    tdm_load_tile_2d((unsigned)(uintptr_t)(void*)sA,
                     agg + (long)row0 * fin, (unsigned)fin, 16u, padc);
    tdm_load_tile_2d((unsigned)(uintptr_t)(void*)sX,
                     xin + (long)row0 * fin, (unsigned)fin, 16u, padc);
    __builtin_amdgcn_s_wait_tensorcnt(0);
  }
  __syncthreads();

  const int ktiles = fin >> 5;
  const int ntf = HID >> 4;                    // fragments per K-tile row
  v8f acc0 = {}, acc1 = {};
  for (int kt = 0; kt < ktiles; ++kt) {
    const int k0 = kt << 5;
    v16bf aA, aX;
#pragma unroll
    for (int e = 0; e < 8; ++e) {
      aA[e]     = f2bf(sA[m * stride + k0 + 8 * h + e]);
      aA[8 + e] = f2bf(sA[m * stride + k0 + 16 + 8 * h + e]);
      aX[e]     = f2bf(sX[m * stride + k0 + 8 * h + e]);
      aX[8 + e] = f2bf(sX[m * stride + k0 + 16 + 8 * h + e]);
    }
    const int fr = kt * ntf + (n0 >> 4);
    v16bf b0 = load_b_packed(WrelP,  fr,     lane);
    v16bf b1 = load_b_packed(WrelP,  fr + 1, lane);
    v16bf c0 = load_b_packed(WrootP, fr,     lane);
    v16bf c1 = load_b_packed(WrootP, fr + 1, lane);
    acc0 = wmma_bf16(aA, b0, acc0);
    acc0 = wmma_bf16(aX, c0, acc0);
    acc1 = wmma_bf16(aA, b1, acc1);
    acc1 = wmma_bf16(aX, c1, acc1);
  }
#pragma unroll
  for (int r = 0; r < 8; ++r) {
    const int mm = row0 + r + 8 * h;
    const int na = n0 + (lane & 15);
    float v0 = acc0[r] + brel[na];
    v0 = fmaxf(v0, 0.f);
    v0 = (v0 - mean[na]) * (gamma[na] * rsqrtf(var[na] + EPSBN)) + beta[na];
    hout[(long)mm * HID + na] = v0;
    const int nb = na + 16;
    float v1 = acc1[r] + brel[nb];
    v1 = fmaxf(v1, 0.f);
    v1 = (v1 - mean[nb]) * (gamma[nb] * rsqrtf(var[nb] + EPSBN)) + beta[nb];
    hout[(long)mm * HID + nb] = v1;
  }
}

// ---------------- p / ||p|| ----------------
__global__ __launch_bounds__(128) void pnorm_kernel(const float* __restrict__ p,
                                                    float* __restrict__ punit) {
  __shared__ float red[128];
  const int t = threadIdx.x;
  const float v = p[t];
  red[t] = v * v;
  __syncthreads();
  for (int off = 64; off > 0; off >>= 1) {
    if (t < off) red[t] += red[t + off];
    __syncthreads();
  }
  punit[t] = v * rsqrtf(red[0]);
}

// ---------------- score = h @ punit ----------------
__global__ void score_kernel(const float* __restrict__ h,
                             const float* __restrict__ punit,
                             float* __restrict__ score) {
  const int v = blockIdx.x * blockDim.x + threadIdx.x;
  if (v >= NTOT) return;
  const float* row = h + (long)v * HID;
  float s = 0.f;
#pragma unroll 8
  for (int f = 0; f < HID; ++f) s += row[f] * punit[f];
  score[v] = s;
}

// ---------------- per-graph top-k (value desc, index asc tiebreak) ----------------
__global__ __launch_bounds__(128) void topk_kernel(const float* __restrict__ score,
                                                   const int* __restrict__ mask_in,
                                                   int k, int* __restrict__ mask_out) {
  __shared__ float s[128];
  __shared__ float rv[128];
  __shared__ int   ri[128];
  const int g = blockIdx.x, t = threadIdx.x, v = g * NNODE + t;
  s[t] = mask_in[v] ? score[v] : -1e30f;
  int sel = 0;
  for (int it = 0; it < k; ++it) {
    __syncthreads();
    rv[t] = s[t]; ri[t] = t;
    __syncthreads();
    for (int off = 64; off > 0; off >>= 1) {
      if (t < off) {
        if (rv[t + off] > rv[t] ||
            (rv[t + off] == rv[t] && ri[t + off] < ri[t])) {
          rv[t] = rv[t + off]; ri[t] = ri[t + off];
        }
      }
      __syncthreads();
    }
    const int win = ri[0];
    __syncthreads();
    if (t == win) { sel = 1; s[t] = -1e30f; }
  }
  mask_out[v] = sel;
}

// ---------------- gate: h = sel ? h*tanh(score) : 0 ----------------
__global__ void gate_kernel(float* __restrict__ h, const int* __restrict__ mask,
                            const float* __restrict__ score, long n) {
  const long idx = blockIdx.x * 256L + threadIdx.x;
  if (idx >= n) return;
  const int v = (int)(idx >> 7);
  h[idx] = mask[v] ? h[idx] * tanhf(score[v]) : 0.f;
}

// ---------------- edge_mask &= sel[src] & sel[dst] ----------------
__global__ void edge_update_kernel(int* __restrict__ emask,
                                   const int* __restrict__ mask,
                                   const int* __restrict__ src,
                                   const int* __restrict__ dst) {
  const int e = blockIdx.x * blockDim.x + threadIdx.x;
  if (e >= EDG) return;
  if (emask[e]) emask[e] = mask[src[e]] & mask[dst[e]];
}

// ---------------- readout += [gmax | gmean] per graph ----------------
__global__ __launch_bounds__(128) void readout_kernel(const float* __restrict__ h,
                                                      const int* __restrict__ mask,
                                                      float kinv,
                                                      float* __restrict__ readout) {
  const int g = blockIdx.x, f = threadIdx.x;
  float mx = -1e30f, sm = 0.f;
  const float* base = h + (long)g * NNODE * HID + f;
  const int* mrow = mask + g * NNODE;
  for (int n = 0; n < NNODE; ++n) {
    if (mrow[n]) {
      const float v = base[(long)n * HID];
      mx = fmaxf(mx, v);
      sm += v;
    }
  }
  readout[g * 2 * HID + f] += mx;
  readout[g * 2 * HID + HID + f] += sm * kinv;
}

// ---------------- generic WMMA GEMM for the MLP head ----------------
// One wave computes a 16x32 tile of C = [relu](A@W + b); packed bf16 B.
__global__ __launch_bounds__(32) void mlp_gemm_kernel(
    const float* __restrict__ A, const __bf16* __restrict__ WP,
    const float* __restrict__ bias, float* __restrict__ C,
    int K, int N, int do_relu) {
  const int row0 = blockIdx.x * 16;
  const int n0 = blockIdx.y * 32;
  const int lane = threadIdx.x & 31;
  const int h = lane >> 4;
  const int ntf = N >> 4;
  v8f acc0 = {}, acc1 = {};
  for (int kt = 0; kt < (K >> 5); ++kt) {
    v16bf a  = load_a_frag_g(A, K, row0, kt << 5);
    const int fr = kt * ntf + (n0 >> 4);
    v16bf b0 = load_b_packed(WP, fr,     lane);
    v16bf b1 = load_b_packed(WP, fr + 1, lane);
    acc0 = wmma_bf16(a, b0, acc0);
    acc1 = wmma_bf16(a, b1, acc1);
  }
#pragma unroll
  for (int r = 0; r < 8; ++r) {
    const int mm = row0 + r + 8 * h;
    const int na = n0 + (lane & 15);
    float v0 = acc0[r] + bias[na];
    if (do_relu) v0 = fmaxf(v0, 0.f);
    C[(long)mm * N + na] = v0;
    const int nb = na + 16;
    float v1 = acc1[r] + bias[nb];
    if (do_relu) v1 = fmaxf(v1, 0.f);
    C[(long)mm * N + nb] = v1;
  }
}

// ---------------- lin3 + log_softmax (NCLS=10) ----------------
__global__ void head_kernel(const float* __restrict__ z, const float* __restrict__ W,
                            const float* __restrict__ b, float* __restrict__ out) {
  const int g = blockIdx.x * blockDim.x + threadIdx.x;
  if (g >= BGR) return;
  float logit[10];
  const float* zr = z + g * 64;
#pragma unroll
  for (int c = 0; c < 10; ++c) logit[c] = b[c];
  for (int f = 0; f < 64; ++f) {
    const float zv = zr[f];
#pragma unroll
    for (int c = 0; c < 10; ++c) logit[c] += zv * W[f * 10 + c];
  }
  float m = logit[0];
#pragma unroll
  for (int c = 1; c < 10; ++c) m = fmaxf(m, logit[c]);
  float s = 0.f;
#pragma unroll
  for (int c = 0; c < 10; ++c) s += expf(logit[c] - m);
  const float lse = logf(s);
#pragma unroll
  for (int c = 0; c < 10; ++c) out[g * 10 + c] = logit[c] - m - lse;
}

extern "C" void kernel_launch(void* const* d_in, const int* in_sizes, int n_in,
                              void* d_out, int out_size, void* d_ws, size_t ws_size,
                              hipStream_t stream) {
  const float* x   = (const float*)d_in[0];
  const int* esrc  = (const int*)d_in[1];
  const int* edst  = (const int*)d_in[2];

  // Workspace carving (~200 MB)
  float* ws      = (float*)d_ws;
  float* hA      = ws;
  float* hBuf    = hA + (size_t)NTOT * HID;
  float* agg     = hBuf + (size_t)NTOT * HID;
  float* score   = agg + (size_t)NTOT * HID;
  int*   nmA     = (int*)(score + NTOT);
  int*   nmB     = nmA + NTOT;
  int*   em      = nmB + NTOT;
  float* readout = (float*)(em + EDG);
  float* punit   = readout + (size_t)BGR * 2 * HID;
  float* z1      = punit + 5 * HID;
  float* z2      = z1 + (size_t)BGR * HID;
  // packed bf16 weights (aligned to 64B)
  uintptr_t pk = (uintptr_t)(z2 + (size_t)BGR * 64);
  pk = (pk + 63) & ~(uintptr_t)63;
  __bf16* wpack = (__bf16*)pk;

  auto blocks = [](long n) { return (unsigned)((n + 255) / 256); };

  // Re-init all mutable state every call (deterministic launch)
  fill_i32<<<blocks(NTOT), 256, 0, stream>>>(nmA, 1, NTOT);
  fill_i32<<<blocks(EDG), 256, 0, stream>>>(em, 1, EDG);
  fill_f32<<<blocks((long)BGR * 2 * HID), 256, 0, stream>>>(readout, 0.f,
                                                            (long)BGR * 2 * HID);

  // Pack all weights to fragment-major bf16 up front
  __bf16* wrelp[5];
  __bf16* wrootp[5];
  __bf16* cur = wpack;
  for (int i = 0; i < 5; ++i) {
    const int fin = (i == 0) ? FIN0 : HID;
    const int base = 3 + 8 * i;
    wrelp[i] = cur;  cur += (size_t)fin * HID;
    wrootp[i] = cur; cur += (size_t)fin * HID;
    const long thr = (long)(fin >> 5) * (HID >> 4) * 32;
    pack_b_kernel<<<blocks(thr), 256, 0, stream>>>((const float*)d_in[base + 0],
                                                   fin, HID, wrelp[i]);
    pack_b_kernel<<<blocks(thr), 256, 0, stream>>>((const float*)d_in[base + 2],
                                                   fin, HID, wrootp[i]);
  }
  __bf16* W1p = cur; cur += (size_t)(2 * HID) * HID;
  __bf16* W2p = cur; cur += (size_t)HID * 64;
  pack_b_kernel<<<blocks((long)(2 * HID >> 5) * (HID >> 4) * 32), 256, 0, stream>>>(
      (const float*)d_in[43], 2 * HID, HID, W1p);
  pack_b_kernel<<<blocks((long)(HID >> 5) * (64 >> 4) * 32), 256, 0, stream>>>(
      (const float*)d_in[45], HID, 64, W2p);

  static const int KS_[5] = {64, 32, 16, 8, 4};
  const float* hin = x;
  float* hout = hA;
  int* min_ = nmA;
  int* mout_ = nmB;

  for (int i = 0; i < 5; ++i) {
    const int fin = (i == 0) ? FIN0 : HID;
    const int base = 3 + 8 * i;
    const float* brel  = (const float*)d_in[base + 1];
    const float* pvec  = (const float*)d_in[base + 3];
    const float* gamma = (const float*)d_in[base + 4];
    const float* beta  = (const float*)d_in[base + 5];
    const float* meanp = (const float*)d_in[base + 6];
    const float* varp  = (const float*)d_in[base + 7];

    gather_kernel<<<BGR, 128, (size_t)NNODE * fin * sizeof(float), stream>>>(
        hin, esrc, edst, em, fin, agg);
    stage_gemm_kernel<<<NTOT / 16, 128, 0, stream>>>(
        agg, hin, fin, wrelp[i], wrootp[i], brel, gamma, beta, meanp, varp, hout);
    pnorm_kernel<<<1, 128, 0, stream>>>(pvec, punit + i * HID);
    score_kernel<<<blocks(NTOT), 256, 0, stream>>>(hout, punit + i * HID, score);
    topk_kernel<<<BGR, 128, 0, stream>>>(score, min_, KS_[i], mout_);
    gate_kernel<<<blocks((long)NTOT * HID), 256, 0, stream>>>(
        hout, mout_, score, (long)NTOT * HID);
    edge_update_kernel<<<blocks(EDG), 256, 0, stream>>>(em, mout_, esrc, edst);
    readout_kernel<<<BGR, 128, 0, stream>>>(hout, mout_, 1.0f / KS_[i], readout);

    hin = hout;
    hout = (hout == hA) ? hBuf : hA;
    int* tmp = min_; min_ = mout_; mout_ = tmp;
  }

  const float* b1 = (const float*)d_in[44];
  const float* b2 = (const float*)d_in[46];
  const float* W3 = (const float*)d_in[47];
  const float* b3 = (const float*)d_in[48];

  dim3 g1(BGR / 16, HID / 32);          // 64 x 4 tiles
  mlp_gemm_kernel<<<g1, 32, 0, stream>>>(readout, W1p, b1, z1, 2 * HID, HID, 1);
  dim3 g2(BGR / 16, 64 / 32);           // 64 x 2 tiles
  mlp_gemm_kernel<<<g2, 32, 0, stream>>>(z1, W2p, b2, z2, HID, 64, 1);
  head_kernel<<<blocks(BGR), 256, 0, stream>>>(z2, W3, b3, (float*)d_out);
}
